// PrunedCrossAttentionBlock_30597347017099
// MI455X (gfx1250) — matrix operand order repaired and checked by reading, not compiled, not run on hardware
//
#include <hip/hip_runtime.h>
#include <hip/hip_bf16.h>

// ---------------- constants (match reference) ----------------
#define BB   4
#define CC   256
#define LL   4096          // HS*WS
#define EE   256
#define NHH  8
#define HDD  32
#define TOPKK 512
#define MLPP 1024
#define EPSF 1e-5f

typedef __attribute__((ext_vector_type(16))) __bf16 v16bf;
typedef __attribute__((ext_vector_type(8)))  __bf16 v8bf;
typedef __attribute__((ext_vector_type(8)))  float  v8f;

static __device__ __forceinline__ __bf16 f2bf(float f) {
    unsigned u = __float_as_uint(f);
    u += 0x7FFFu + ((u >> 16) & 1u);        // round-to-nearest-even
    unsigned short h = (unsigned short)(u >> 16);
    __bf16 r;
    __builtin_memcpy(&r, &h, 2);
    return r;
}

static __device__ __forceinline__ v16bf mk16(v8bf lo, v8bf hi) {
    return __builtin_shufflevector(lo, hi, 0,1,2,3,4,5,6,7,8,9,10,11,12,13,14,15);
}

// ---------------- small utility kernels ----------------

__global__ void cvt_f32_to_bf16(const float* __restrict__ src,
                                __bf16* __restrict__ dst, int n) {
    int i = blockIdx.x * blockDim.x + threadIdx.x;
    if (i < n) dst[i] = f2bf(src[i]);
}

// scale = gamma/sqrt(var+eps); shift = conv_b*scale + beta - mean*scale
__global__ void bn_prep(const float* __restrict__ g, const float* __restrict__ beta,
                        const float* __restrict__ mean, const float* __restrict__ var,
                        const float* __restrict__ cb,
                        float* __restrict__ scale, float* __restrict__ shift, int n) {
    int i = blockIdx.x * blockDim.x + threadIdx.x;
    if (i < n) {
        float s = g[i] * rsqrtf(var[i] + EPSF);
        scale[i] = s;
        shift[i] = cb[i] * s + beta[i] - mean[i] * s;
    }
}

// [B,C,L] f32 -> [B,L,C] bf16  (tiled transpose)
__global__ __launch_bounds__(256) void transpose_cvt(const float* __restrict__ in,
                                                     __bf16* __restrict__ out) {
    __shared__ float tile[32][33];
    int b  = blockIdx.z;
    int l0 = blockIdx.x * 32;
    int c0 = blockIdx.y * 32;
    int tx = threadIdx.x & 31;
    int ty = threadIdx.x >> 5;     // 0..7
    #pragma unroll
    for (int rr = 0; rr < 32; rr += 8) {
        int c = c0 + ty + rr;
        tile[ty + rr][tx] = in[((size_t)b * CC + c) * LL + l0 + tx];
    }
    __syncthreads();
    #pragma unroll
    for (int rr = 0; rr < 32; rr += 8) {
        int l = l0 + ty + rr;
        out[((size_t)b * LL + l) * CC + c0 + tx] = f2bf(tile[tx][ty + rr]);
    }
}

// ---------------- stage 1: min L1 distance ----------------
__global__ __launch_bounds__(256) void min_l1_kernel(const float* __restrict__ qsrc,
                                                     const float* __restrict__ ctx,
                                                     const int* __restrict__ rind,
                                                     float* __restrict__ md) {
    __shared__ float Q[16 * 256];
    int b = blockIdx.y;
    int l = blockIdx.x * 256 + threadIdx.x;
    const float* ctxb = ctx + (size_t)b * CC * LL;
    float best = 3.4e38f;
    for (int qc = 0; qc < 32; ++qc) {
        __syncthreads();
        #pragma unroll
        for (int i = 0; i < 16; ++i) {
            int c = threadIdx.x;
            int tok = rind[b * TOPKK + qc * 16 + i];
            Q[i * 256 + c] = qsrc[((size_t)b * CC + c) * LL + tok];
        }
        __syncthreads();
        float dist[16];
        #pragma unroll
        for (int i = 0; i < 16; ++i) dist[i] = 0.f;
        for (int c = 0; c < 256; ++c) {
            float kv = ctxb[(size_t)c * LL + l];   // coalesced
            #pragma unroll
            for (int i = 0; i < 16; ++i)
                dist[i] += fabsf(kv - Q[i * 256 + c]);
        }
        #pragma unroll
        for (int i = 0; i < 16; ++i) best = fminf(best, dist[i]);
    }
    md[(size_t)b * LL + l] = best;
}

// ---------------- stage 2: per-batch bitonic sort, take 512 smallest ----------------
__global__ __launch_bounds__(1024) void topk_sort(const float* __restrict__ md,
                                                  int* __restrict__ topidx) {
    __shared__ float key[4096];
    __shared__ int   idx[4096];
    int b = blockIdx.x;
    for (int i = threadIdx.x; i < 4096; i += 1024) { key[i] = md[(size_t)b * LL + i]; idx[i] = i; }
    __syncthreads();
    for (int k2 = 2; k2 <= 4096; k2 <<= 1) {
        for (int j = k2 >> 1; j > 0; j >>= 1) {
            for (int i = threadIdx.x; i < 4096; i += 1024) {
                int ixj = i ^ j;
                if (ixj > i) {
                    bool up = ((i & k2) == 0);
                    float ka = key[i], kb = key[ixj];
                    int   ia = idx[i], ib = idx[ixj];
                    bool gt = (ka > kb) || (ka == kb && ia > ib);  // stable tie-break
                    if (gt == up) { key[i] = kb; key[ixj] = ka; idx[i] = ib; idx[ixj] = ia; }
                }
            }
            __syncthreads();
        }
    }
    for (int i = threadIdx.x; i < TOPKK; i += 1024) topidx[b * TOPKK + i] = idx[i];
}

// gather selected kv rows -> bf16 [B*512, 256]
__global__ __launch_bounds__(256) void gather_ksel(const float* __restrict__ ctx,
                                                   const int* __restrict__ topidx,
                                                   __bf16* __restrict__ ksel) {
    int row = blockIdx.x;         // b*512 + t
    int b = row >> 9;
    int c = threadIdx.x;
    int tok = topidx[row];
    ksel[(size_t)row * CC + c] = f2bf(ctx[((size_t)b * CC + c) * LL + tok]);
}

// ---------------- generic WMMA GEMM: D = A[M,K] * Bt[N,K]^T ----------------
// Register-blocked: each wave computes a 32(M) x 64(N) tile -> 8 WMMAs per K-slab
// for 4 A-chunk loads + 8 B-chunk loads (1.5 b128 loads per WMMA).
// modes: 0 bias->bf16 ; 1 bias->v_t layout ; 2 bias->f32+bf16 ;
//        3 bn+relu->bf16 ; 4 bn+residual -> NCHW f32 (final output)
__global__ __launch_bounds__(128) void gemm_bf16(const __bf16* __restrict__ A,
                                                 const __bf16* __restrict__ Bt,
                                                 int M, int N, int K,
                                                 const float* __restrict__ bias,
                                                 const float* __restrict__ scale,
                                                 const float* __restrict__ shift,
                                                 __bf16* __restrict__ outBf,
                                                 float* __restrict__ outF,
                                                 const float* __restrict__ resF,
                                                 float* __restrict__ finalOut,
                                                 int mode) {
    int wave = threadIdx.x >> 5;
    int lane = threadIdx.x & 31;
    int m0 = blockIdx.x * 32;
    int n0 = (blockIdx.y * 4 + wave) * 64;
    if (m0 >= M || n0 >= N) return;
    int mrow = lane & 15;
    int hi   = lane >> 4;

    const __bf16* Arow0 = A + (size_t)(m0 + mrow) * K;
    const __bf16* Arow1 = A + (size_t)(m0 + 16 + mrow) * K;
    const __bf16* Brow0 = Bt + (size_t)(n0 +  0 + mrow) * K;
    const __bf16* Brow1 = Bt + (size_t)(n0 + 16 + mrow) * K;
    const __bf16* Brow2 = Bt + (size_t)(n0 + 32 + mrow) * K;
    const __bf16* Brow3 = Bt + (size_t)(n0 + 48 + mrow) * K;

    v8f acc[2][4] = {};
    for (int kk = 0; kk < K; kk += 32) {
        // A 16x32 tiles: lane(M=mrow) elements K=e+16*hi
        v16bf a0 = mk16(*(const v8bf*)(Arow0 + kk + 16 * hi),
                        *(const v8bf*)(Arow0 + kk + 16 * hi + 8));
        v16bf a1 = mk16(*(const v8bf*)(Arow1 + kk + 16 * hi),
                        *(const v8bf*)(Arow1 + kk + 16 * hi + 8));
        // B 32x16 tiles: lane(N=mrow) elements K=e+16*hi
        v16bf b0 = mk16(*(const v8bf*)(Brow0 + kk + 16 * hi),
                        *(const v8bf*)(Brow0 + kk + 16 * hi + 8));
        v16bf b1 = mk16(*(const v8bf*)(Brow1 + kk + 16 * hi),
                        *(const v8bf*)(Brow1 + kk + 16 * hi + 8));
        v16bf b2 = mk16(*(const v8bf*)(Brow2 + kk + 16 * hi),
                        *(const v8bf*)(Brow2 + kk + 16 * hi + 8));
        v16bf b3 = mk16(*(const v8bf*)(Brow3 + kk + 16 * hi),
                        *(const v8bf*)(Brow3 + kk + 16 * hi + 8));
        acc[0][0] = __builtin_amdgcn_wmma_f32_16x16x32_bf16(false, a0, false, b0, (short)0, acc[0][0], false, false);
        acc[1][0] = __builtin_amdgcn_wmma_f32_16x16x32_bf16(false, a1, false, b0, (short)0, acc[1][0], false, false);
        acc[0][1] = __builtin_amdgcn_wmma_f32_16x16x32_bf16(false, a0, false, b1, (short)0, acc[0][1], false, false);
        acc[1][1] = __builtin_amdgcn_wmma_f32_16x16x32_bf16(false, a1, false, b1, (short)0, acc[1][1], false, false);
        acc[0][2] = __builtin_amdgcn_wmma_f32_16x16x32_bf16(false, a0, false, b2, (short)0, acc[0][2], false, false);
        acc[1][2] = __builtin_amdgcn_wmma_f32_16x16x32_bf16(false, a1, false, b2, (short)0, acc[1][2], false, false);
        acc[0][3] = __builtin_amdgcn_wmma_f32_16x16x32_bf16(false, a0, false, b3, (short)0, acc[0][3], false, false);
        acc[1][3] = __builtin_amdgcn_wmma_f32_16x16x32_bf16(false, a1, false, b3, (short)0, acc[1][3], false, false);
    }

    #pragma unroll
    for (int t = 0; t < 4; ++t) {
        int n = n0 + t * 16 + mrow;      // output column for this lane
        float bsc = 0.f, bsh = 0.f;
        if (mode == 3 || mode == 4) { bsc = scale[n]; bsh = shift[n]; }
        else                        { bsh = bias[n]; }
        #pragma unroll
        for (int mi = 0; mi < 2; ++mi) {
            #pragma unroll
            for (int r = 0; r < 8; ++r) {
                int m = m0 + mi * 16 + r + 8 * hi;   // output row
                float v = acc[mi][t][r];
                if (mode == 0) {
                    outBf[(size_t)m * N + n] = f2bf(v + bsh);
                } else if (mode == 1) {
                    v += bsh;
                    int bb = m >> 9, tt = m & 511;   // rows are b*512+t
                    int h = n >> 5,  d = n & 31;     // cols are h*32+d
                    outBf[(((size_t)bb * NHH + h) * HDD + d) * TOPKK + tt] = f2bf(v);
                } else if (mode == 2) {
                    v += bsh;
                    outF [(size_t)m * N + n] = v;
                    outBf[(size_t)m * N + n] = f2bf(v);
                } else if (mode == 3) {
                    v = v * bsc + bsh;
                    v = v > 0.f ? v : 0.f;
                    outBf[(size_t)m * N + n] = f2bf(v);
                } else {
                    v = v * bsc + bsh;
                    float res = resF[(size_t)m * EE + n] + v;
                    int bb = m >> 12, l = m & 4095;
                    finalOut[((size_t)bb * CC + n) * LL + l] = res;
                }
            }
        }
    }
}

// ---------------- fused attention: per (b,h,16-row q tile) ----------------
__global__ __launch_bounds__(32) void attn_kernel(const __bf16* __restrict__ qbf,
                                                  const __bf16* __restrict__ kbf,
                                                  const __bf16* __restrict__ vt,
                                                  __bf16* __restrict__ ctxbf) {
    __shared__ __align__(16) float  S[16][512];
    __shared__ __align__(16) __bf16 P[16][512];
    int lane = threadIdx.x & 31;
    int qt = blockIdx.x;   // 0..255
    int h  = blockIdx.y;   // 0..7
    int b  = blockIdx.z;   // 0..3
    int mrow = lane & 15;
    int hi   = lane >> 4;

    // A tile: q rows [qt*16 .. +16), head slice h*32..+32
    const __bf16* Arow = qbf + ((size_t)(b * LL + qt * 16 + mrow)) * EE + h * HDD;
    v16bf a = mk16(*(const v8bf*)(Arow + 16 * hi),
                   *(const v8bf*)(Arow + 16 * hi + 8));

    const float sc = 0.17677669529663687f;   // 1/sqrt(HD)
    for (int kt = 0; kt < 32; ++kt) {
        const __bf16* Brow = kbf + ((size_t)(b * TOPKK + kt * 16 + mrow)) * EE + h * HDD + 16 * hi;
        v16bf bb = mk16(*(const v8bf*)(Brow), *(const v8bf*)(Brow + 8));
        v8f d = {};
        d = __builtin_amdgcn_wmma_f32_16x16x32_bf16(false, a, false, bb, (short)0, d, false, false);
        #pragma unroll
        for (int r = 0; r < 8; ++r)
            S[r + 8 * hi][kt * 16 + mrow] = d[r] * sc;
    }
    __syncthreads();

    if (lane < 16) {
        float mx = -3.4e38f;
        for (int j = 0; j < 512; ++j) mx = fmaxf(mx, S[lane][j]);
        float sum = 0.f;
        for (int j = 0; j < 512; ++j) { float e = __expf(S[lane][j] - mx); S[lane][j] = e; sum += e; }
        float inv = 1.0f / sum;
        for (int j = 0; j < 512; ++j) P[lane][j] = f2bf(S[lane][j] * inv);
    }
    __syncthreads();

    v8f o0 = {}, o1 = {};
    const __bf16* Vh = vt + ((size_t)(b * NHH + h)) * HDD * TOPKK;
    for (int slab = 0; slab < 16; ++slab) {
        const __bf16* Prow = &P[mrow][slab * 32];
        v16bf ap = mk16(*(const v8bf*)(Prow + 16 * hi),
                        *(const v8bf*)(Prow + 16 * hi + 8));
        const __bf16* V0 = Vh + ((size_t)mrow)        * TOPKK + slab * 32 + 16 * hi;
        const __bf16* V1 = Vh + ((size_t)(16 + mrow)) * TOPKK + slab * 32 + 16 * hi;
        v16bf bv0 = mk16(*(const v8bf*)(V0), *(const v8bf*)(V0 + 8));
        v16bf bv1 = mk16(*(const v8bf*)(V1), *(const v8bf*)(V1 + 8));
        o0 = __builtin_amdgcn_wmma_f32_16x16x32_bf16(false, ap, false, bv0, (short)0, o0, false, false);
        o1 = __builtin_amdgcn_wmma_f32_16x16x32_bf16(false, ap, false, bv1, (short)0, o1, false, false);
    }

    #pragma unroll
    for (int r = 0; r < 8; ++r) {
        int q = qt * 16 + r + 8 * hi;
        size_t base = ((size_t)(b * LL + q)) * EE + h * HDD;
        ctxbf[base + mrow]      = f2bf(o0[r]);
        ctxbf[base + 16 + mrow] = f2bf(o1[r]);
    }
}

// ---------------- host launcher ----------------
extern "C" void kernel_launch(void* const* d_in, const int* in_sizes, int n_in,
                              void* d_out, int out_size, void* d_ws, size_t ws_size,
                              hipStream_t stream) {
    const float* qsrc  = (const float*)d_in[0];
    const float* ctx   = (const float*)d_in[1];
    const int*   rind  = (const int*)  d_in[2];
    const float* ipw   = (const float*)d_in[3];   // [768,256]
    const float* ipb   = (const float*)d_in[4];   // [768]
    const float* opw   = (const float*)d_in[5];   // [256,256]
    const float* opb   = (const float*)d_in[6];   // [256]
    const float* c1w   = (const float*)d_in[7];   // [1024,256]
    const float* c1b   = (const float*)d_in[8];
    const float* g1    = (const float*)d_in[9];
    const float* be1   = (const float*)d_in[10];
    const float* mn1   = (const float*)d_in[11];
    const float* vr1   = (const float*)d_in[12];
    const float* c2w   = (const float*)d_in[13];  // [256,1024]
    const float* c2b   = (const float*)d_in[14];
    const float* g2    = (const float*)d_in[15];
    const float* be2   = (const float*)d_in[16];
    const float* mn2   = (const float*)d_in[17];
    const float* vr2   = (const float*)d_in[18];
    float* out = (float*)d_out;

    // workspace partition (256B aligned)
    char* base = (char*)d_ws;
    size_t off = 0;
    auto alloc = [&](size_t bytes) -> char* {
        char* p = base + off;
        off += (bytes + 255) & ~(size_t)255;
        return p;
    };
    float*  md      = (float*) alloc((size_t)BB * LL * 4);
    int*    topidx  = (int*)   alloc((size_t)BB * TOPKK * 4);
    __bf16* wq_bf   = (__bf16*)alloc((size_t)EE * EE * 2);
    __bf16* wk_bf   = (__bf16*)alloc((size_t)EE * EE * 2);
    __bf16* wv_bf   = (__bf16*)alloc((size_t)EE * EE * 2);
    __bf16* ow_bf   = (__bf16*)alloc((size_t)EE * EE * 2);
    __bf16* w1_bf   = (__bf16*)alloc((size_t)MLPP * CC * 2);
    __bf16* w2_bf   = (__bf16*)alloc((size_t)CC * MLPP * 2);
    float*  s1      = (float*) alloc(MLPP * 4);
    float*  sh1     = (float*) alloc(MLPP * 4);
    float*  s2      = (float*) alloc(CC * 4);
    float*  sh2     = (float*) alloc(CC * 4);
    __bf16* qtok_bf = (__bf16*)alloc((size_t)BB * LL * CC * 2);
    __bf16* ksel_bf = (__bf16*)alloc((size_t)BB * TOPKK * CC * 2);
    __bf16* q_bf    = (__bf16*)alloc((size_t)BB * LL * EE * 2);
    __bf16* k_bf    = (__bf16*)alloc((size_t)BB * TOPKK * EE * 2);
    __bf16* v_t     = (__bf16*)alloc((size_t)BB * NHH * HDD * TOPKK * 2);
    __bf16* ctx_bf  = (__bf16*)alloc((size_t)BB * LL * EE * 2);
    float*  attn_f  = (float*) alloc((size_t)BB * LL * EE * 4);
    __bf16* attn_bf = (__bf16*)alloc((size_t)BB * LL * EE * 2);
    __bf16* hdn_bf  = (__bf16*)alloc((size_t)BB * LL * MLPP * 2);

    // 1. convert weights to bf16
    cvt_f32_to_bf16<<<256, 256, 0, stream>>>(ipw,               wq_bf, EE * EE);
    cvt_f32_to_bf16<<<256, 256, 0, stream>>>(ipw + EE * EE,     wk_bf, EE * EE);
    cvt_f32_to_bf16<<<256, 256, 0, stream>>>(ipw + 2 * EE * EE, wv_bf, EE * EE);
    cvt_f32_to_bf16<<<256, 256, 0, stream>>>(opw,               ow_bf, EE * EE);
    cvt_f32_to_bf16<<<1024, 256, 0, stream>>>(c1w,              w1_bf, MLPP * CC);
    cvt_f32_to_bf16<<<1024, 256, 0, stream>>>(c2w,              w2_bf, CC * MLPP);

    // 2. fold BN
    bn_prep<<<4, 256, 0, stream>>>(g1, be1, mn1, vr1, c1b, s1, sh1, MLPP);
    bn_prep<<<1, 256, 0, stream>>>(g2, be2, mn2, vr2, c2b, s2, sh2, CC);

    // 3. [B,C,L] -> [B,L,C] bf16
    transpose_cvt<<<dim3(LL / 32, CC / 32, BB), 256, 0, stream>>>(qsrc, qtok_bf);

    // 4. min L1 distance
    min_l1_kernel<<<dim3(LL / 256, BB), 256, 0, stream>>>(qsrc, ctx, rind, md);

    // 5. top-512 smallest per batch (bitonic, stable tie-break)
    topk_sort<<<BB, 1024, 0, stream>>>(md, topidx);

    // 6. gather selected kv rows
    gather_ksel<<<BB * TOPKK, 256, 0, stream>>>(ctx, topidx, ksel_bf);

    // 7. QKV projections (WMMA, 32x64 per wave)
    gemm_bf16<<<dim3(BB * LL / 32, EE / 256), 128, 0, stream>>>(
        qtok_bf, wq_bf, BB * LL, EE, EE, ipb, nullptr, nullptr,
        q_bf, nullptr, nullptr, nullptr, 0);
    gemm_bf16<<<dim3(BB * TOPKK / 32, EE / 256), 128, 0, stream>>>(
        ksel_bf, wk_bf, BB * TOPKK, EE, EE, ipb + EE, nullptr, nullptr,
        k_bf, nullptr, nullptr, nullptr, 0);
    gemm_bf16<<<dim3(BB * TOPKK / 32, EE / 256), 128, 0, stream>>>(
        ksel_bf, wv_bf, BB * TOPKK, EE, EE, ipb + 2 * EE, nullptr, nullptr,
        v_t, nullptr, nullptr, nullptr, 1);

    // 8. fused attention (WMMA + LDS softmax)
    attn_kernel<<<dim3(LL / 16, NHH, BB), 32, 0, stream>>>(q_bf, k_bf, v_t, ctx_bf);

    // 9. out projection -> attn_out (f32 + bf16)
    gemm_bf16<<<dim3(BB * LL / 32, EE / 256), 128, 0, stream>>>(
        ctx_bf, ow_bf, BB * LL, EE, EE, opb, nullptr, nullptr,
        attn_bf, attn_f, nullptr, nullptr, 2);

    // 10. conv1 + BN1 + ReLU
    gemm_bf16<<<dim3(BB * LL / 32, MLPP / 256), 128, 0, stream>>>(
        attn_bf, w1_bf, BB * LL, MLPP, CC, nullptr, s1, sh1,
        hdn_bf, nullptr, nullptr, nullptr, 3);

    // 11. conv2 + BN2 + residual -> NCHW output
    gemm_bf16<<<dim3(BB * LL / 32, EE / 256), 128, 0, stream>>>(
        hdn_bf, w2_bf, BB * LL, EE, MLPP, nullptr, s2, sh2,
        nullptr, nullptr, attn_f, out, 4);

    (void)in_sizes; (void)n_in; (void)out_size; (void)ws_size;
}
